// SConv_40020505264738
// MI455X (gfx1250) — compile-verified, matrix operand order checked by reading, not run on hardware
//
#include <hip/hip_runtime.h>
#include <hip/hip_bf16.h>
#include <math.h>

typedef __bf16 bf16_t;
typedef __bf16 bf16x8  __attribute__((ext_vector_type(8)));
typedef __bf16 bf16x16 __attribute__((ext_vector_type(16)));
typedef float  f32x8   __attribute__((ext_vector_type(8)));
typedef int    v4i     __attribute__((vector_size(16)));   // 4 x i32

#define DEPTHN 6
#define DIMN   1024
#define FFN    4096
#define VOCABN 32000
#define BBN    2
#define LLN    2048
#define MTOK   (BBN * LLN)   /* 4096 rows */
#define NCH    16            /* sconv chunks */
#define CLEN   (LLN / NCH)   /* 128 steps per chunk */

#define LDS_STRIDE 40        /* bf16 elems per row (32 data + 8 pad = 80B) */

// ---------------------------------------------------------------------------
// CDNA5 async global->LDS copy (16B per lane), with portable fallbacks.
// Builtin signature (probed via diagnostics): (AS1 v4i* src, AS3 v4i* dst,
// imm offset, imm cpol).
// ---------------------------------------------------------------------------
__device__ __forceinline__ void async_copy16(const bf16_t* g, bf16_t* l) {
#if __has_builtin(__builtin_amdgcn_global_load_async_to_lds_b128)
  __builtin_amdgcn_global_load_async_to_lds_b128(
      (__attribute__((address_space(1))) v4i*)(void*)const_cast<bf16_t*>(g),
      (__attribute__((address_space(3))) v4i*)(void*)l, 0, 0);
#else
  *(bf16x8*)l = *(const bf16x8*)g;   // synchronous fallback
#endif
}

template <int N>
__device__ __forceinline__ void async_wait() {
#if __has_builtin(__builtin_amdgcn_global_load_async_to_lds_b128)
#if __has_builtin(__builtin_amdgcn_s_wait_asynccnt)
  __builtin_amdgcn_s_wait_asynccnt(N);
#else
  asm volatile("s_wait_asynccnt %0" ::"i"(N) : "memory");
#endif
#endif
}

// ---------------------------------------------------------------------------
// fp32 -> bf16 conversion (grid-stride)
// ---------------------------------------------------------------------------
__global__ __launch_bounds__(256) void k_cvt_bf16(const float* __restrict__ in,
                                                  bf16_t* __restrict__ out,
                                                  size_t n) {
  size_t i = (size_t)blockIdx.x * blockDim.x + threadIdx.x;
  size_t stride = (size_t)gridDim.x * blockDim.x;
  for (; i < n; i += stride) out[i] = (bf16_t)in[i];
}

// ---------------------------------------------------------------------------
// Embedding gather + max_norm=1 clip. One block per token row.
// ---------------------------------------------------------------------------
__global__ __launch_bounds__(256) void k_embed(const int* __restrict__ tokens,
                                               const float* __restrict__ embed,
                                               float* __restrict__ x) {
  int row = blockIdx.x;
  int tid = threadIdx.x;
  const float* e = embed + (size_t)tokens[row] * DIMN;
  float ss = 0.f;
  for (int i = tid; i < DIMN; i += 256) { float v = e[i]; ss += v * v; }
  __shared__ float red[256];
  red[tid] = ss;
  __syncthreads();
  for (int off = 128; off > 0; off >>= 1) {
    if (tid < off) red[tid] += red[tid + off];
    __syncthreads();
  }
  float n = sqrtf(red[0]);
  float s = (n > 1.0f) ? 1.0f / (n + 1e-7f) : 1.0f;
  float* xr = x + (size_t)row * DIMN;
  for (int i = tid; i < DIMN; i += 256) xr[i] = e[i] * s;
}

// ---------------------------------------------------------------------------
// LayerNorm. One block per row. Emits fp32 (for sconv) + bf16 (for WMMA GEMM).
// ---------------------------------------------------------------------------
__global__ __launch_bounds__(256) void k_layernorm(const float* __restrict__ x,
                                                   const float* __restrict__ w,
                                                   const float* __restrict__ b,
                                                   float* __restrict__ outF,
                                                   bf16_t* __restrict__ outH) {
  int row = blockIdx.x, tid = threadIdx.x;
  const float* xr = x + (size_t)row * DIMN;
  float s = 0.f, s2 = 0.f;
  for (int i = tid; i < DIMN; i += 256) { float v = xr[i]; s += v; s2 += v * v; }
  __shared__ float rs[256];
  __shared__ float rq[256];
  rs[tid] = s; rq[tid] = s2;
  __syncthreads();
  for (int off = 128; off > 0; off >>= 1) {
    if (tid < off) { rs[tid] += rs[tid + off]; rq[tid] += rq[tid + off]; }
    __syncthreads();
  }
  float mean = rs[0] * (1.0f / DIMN);
  float var  = rq[0] * (1.0f / DIMN) - mean * mean;
  float inv  = rsqrtf(var + 1e-5f);
  for (int i = tid; i < DIMN; i += 256) {
    float v = (xr[i] - mean) * inv * w[i] + b[i];
    if (outF) outF[(size_t)row * DIMN + i] = v;
    outH[(size_t)row * DIMN + i] = (bf16_t)v;
  }
}

// ---------------------------------------------------------------------------
// SConv = diagonal complex linear recurrence (equivalent to the FFT causal
// long-conv in the reference):
//   G[t] = phazor * G[t-1] + x[t],  phazor = exp((i*angle - decay) * scale)
//   y[t] = Re( pinit*G[t] + lastc*phazor^{t+1} ) * (1 - exp(-decay*scale))
// Chunked 3-pass scan over L.
// ---------------------------------------------------------------------------
__device__ __forceinline__ void sconv_params(const float* __restrict__ ph_angle,
                                             const float* __restrict__ lnmln,
                                             int d, float& pr, float& pi,
                                             float& dsc, float& th) {
  const float LA = -3.4657359027997265f;  // log(1/32)
  const float LB = -7.6246189861593985f;  // log(1/2048)
  float sc = __expf(LA + (LB - LA) * ((float)d / (float)(DIMN - 1)));
  float decay = __expf(lnmln[d]);
  dsc = decay * sc;
  th = ph_angle[d] * sc;
  float r = __expf(-dsc);
  float c, s;
  __sincosf(th, &s, &c);   // s=sin, c=cos
  pr = r * c;
  pi = r * s;
}

// Pass 1: per-(b,d,chunk) local scan with zero init -> chunk-final state.
__global__ __launch_bounds__(256) void k_sconv_scan(const float* __restrict__ xn,
                                                    const float* __restrict__ ph_angle,
                                                    const float* __restrict__ lnmln,
                                                    float2* __restrict__ cs) {
  int tid = blockIdx.x * blockDim.x + threadIdx.x;
  if (tid >= BBN * NCH * DIMN) return;
  int d = tid % DIMN;
  int c = (tid / DIMN) % NCH;
  int b = tid / (DIMN * NCH);
  float pr, pi, dsc, th;
  sconv_params(ph_angle, lnmln, d, pr, pi, dsc, th);
  float gr = 0.f, gi = 0.f;
  const float* xp = xn + ((size_t)b * LLN + (size_t)c * CLEN) * DIMN + d;
  for (int t = 0; t < CLEN; ++t) {
    float xv = xp[(size_t)t * DIMN];
    float nr = pr * gr - pi * gi + xv;
    float ni = pr * gi + pi * gr;
    gr = nr; gi = ni;
  }
  cs[((size_t)b * DIMN + d) * NCH + c] = make_float2(gr, gi);
}

// Pass 2: per-(b,d) exclusive combine of chunk states with phazor^CLEN.
__global__ __launch_bounds__(256) void k_sconv_combine(const float* __restrict__ ph_angle,
                                                       const float* __restrict__ lnmln,
                                                       float2* __restrict__ cs) {
  int tid = blockIdx.x * blockDim.x + threadIdx.x;
  if (tid >= BBN * DIMN) return;
  int d = tid % DIMN;
  float pr, pi, dsc, th;
  sconv_params(ph_angle, lnmln, d, pr, pi, dsc, th);
  float R = __expf(-dsc * (float)CLEN);
  float sa, ca;
  __sincosf(th * (float)CLEN, &sa, &ca);
  float pcr = R * ca, pci = R * sa;
  float gr = 0.f, gi = 0.f;
  float2* p = cs + (size_t)tid * NCH;
  for (int c = 0; c < NCH; ++c) {
    float2 s = p[c];
    p[c] = make_float2(gr, gi);     // exclusive prefix (incoming state)
    float nr = pcr * gr - pci * gi + s.x;
    float ni = pcr * gi + pci * gr + s.y;
    gr = nr; gi = ni;
  }
}

// Pass 3: rescan with correct incoming state; fuse output scale, gate, residual.
__global__ __launch_bounds__(256) void k_sconv_out(const float* __restrict__ xn,
                                                   const float* __restrict__ gate,
                                                   float* __restrict__ x,
                                                   const float* __restrict__ ph_angle,
                                                   const float* __restrict__ lnmln,
                                                   const float* __restrict__ ph_init,
                                                   const float* __restrict__ last_conv,
                                                   const float2* __restrict__ cs) {
  int tid = blockIdx.x * blockDim.x + threadIdx.x;
  if (tid >= BBN * NCH * DIMN) return;
  int d = tid % DIMN;
  int c = (tid / DIMN) % NCH;
  int b = tid / (DIMN * NCH);
  float pr, pi, dsc, th;
  sconv_params(ph_angle, lnmln, d, pr, pi, dsc, th);
  float pir = ph_init[2 * d], pii = ph_init[2 * d + 1];
  float lr = last_conv[2 * d], li = last_conv[2 * d + 1];
  float ysc = 1.f - __expf(-dsc);
  float2 g0 = cs[((size_t)b * DIMN + d) * NCH + c];
  float gr = g0.x, gi = g0.y;
  for (int t = 0; t < CLEN; ++t) {
    int gt = c * CLEN + t;
    size_t idx = ((size_t)b * LLN + gt) * DIMN + d;
    float xv = xn[idx];
    float nr = pr * gr - pi * gi + xv;
    float ni = pr * gi + pi * gr;
    gr = nr; gi = ni;
    float np1 = (float)(gt + 1);
    float Rn = __expf(-dsc * np1);
    float sk, ck;
    __sincosf(th * np1, &sk, &ck);
    float lre = Rn * (lr * ck - li * sk);
    float y = (pir * gr - pii * gi + lre) * ysc;
    x[idx] = y * gate[idx] + x[idx];
  }
}

// ---------------------------------------------------------------------------
// bf16 WMMA GEMM with double-buffered async global->LDS staging:
//   C[M,N] = A[M,K] @ W[N,K]^T  (+bias, +silu, +residual)
// 256 threads = 8 waves; block tile 128x128; wave tile 32x64 (2x4 WMMA tiles).
// Per K-stage (32): 8KB A-slice + 8KB B-slice copied to LDS with
// GLOBAL_LOAD_ASYNC_TO_LDS_B128 (4 x 16B per thread), rows padded to 80B so
// ds_load_b128 fragment reads are bank-conflict-free.
// ---------------------------------------------------------------------------
union FragAB { bf16x16 v; bf16x8 h[2]; };

__device__ __forceinline__ void stage_copy(const bf16_t* __restrict__ A,
                                           const bf16_t* __restrict__ W,
                                           bf16_t* sAb, bf16_t* sBb,
                                           int mBlk, int nBlk, int K, int k,
                                           int tid) {
#pragma unroll
  for (int r = 0; r < 2; ++r) {
    int t = tid + 256 * r;        // 0..511
    int row = t >> 2;             // 0..127
    int chunk = t & 3;            // 16B chunk within the 64B K-slice row
    async_copy16(A + (size_t)(mBlk + row) * K + k + chunk * 8,
                 sAb + row * LDS_STRIDE + chunk * 8);
    async_copy16(W + (size_t)(nBlk + row) * K + k + chunk * 8,
                 sBb + row * LDS_STRIDE + chunk * 8);
  }
}

__global__ __launch_bounds__(256) void k_gemm_bf16(const bf16_t* __restrict__ A,
                                                   const bf16_t* __restrict__ W,
                                                   const float* __restrict__ bias,
                                                   const float* __restrict__ resid,
                                                   float* __restrict__ outF,
                                                   bf16_t* __restrict__ outH,
                                                   int N, int K, int act) {
  __shared__ bf16_t sA[2][128 * LDS_STRIDE];
  __shared__ bf16_t sB[2][128 * LDS_STRIDE];

  const int tid   = threadIdx.x;
  const int wave  = tid >> 5;
  const int lane  = tid & 31;
  const int waveM = wave >> 1;
  const int waveN = wave & 1;
  const int mBlk  = blockIdx.x * 128;
  const int nBlk  = blockIdx.y * 128;
  const int lrow  = lane & 15;    // A row / B col / C col within 16-tile
  const int kh    = lane >> 4;    // K-half for A/B frags; M-half for C/D

  f32x8 acc[2][4] = {};

  const int NS = K / 32;
  stage_copy(A, W, sA[0], sB[0], mBlk, nBlk, K, 0, tid);

  int buf = 0;
  for (int ks = 0; ks < NS; ++ks) {
    if (ks + 1 < NS) {
      stage_copy(A, W, sA[buf ^ 1], sB[buf ^ 1], mBlk, nBlk, K, (ks + 1) * 32, tid);
      async_wait<4>();   // previous stage (4 older copies) complete
    } else {
      async_wait<0>();
    }
    __syncthreads();

    FragAB a[2], b[4];
    const bf16_t* aB = &sA[buf][0];
    const bf16_t* bB = &sB[buf][0];
#pragma unroll
    for (int i = 0; i < 2; ++i) {
      const bf16_t* p = aB + (waveM * 32 + i * 16 + lrow) * LDS_STRIDE + kh * 8;
      a[i].h[0] = *(const bf16x8*)p;
      a[i].h[1] = *(const bf16x8*)(p + 16);
    }
#pragma unroll
    for (int j = 0; j < 4; ++j) {
      const bf16_t* p = bB + (waveN * 64 + j * 16 + lrow) * LDS_STRIDE + kh * 8;
      b[j].h[0] = *(const bf16x8*)p;
      b[j].h[1] = *(const bf16x8*)(p + 16);
    }
#pragma unroll
    for (int i = 0; i < 2; ++i)
#pragma unroll
      for (int j = 0; j < 4; ++j)
        acc[i][j] = __builtin_amdgcn_wmma_f32_16x16x32_bf16(
            false, a[i].v, false, b[j].v, (short)0, acc[i][j], false, false);

    __syncthreads();   // protect buffer reuse by the next stage's copies
    buf ^= 1;
  }

#pragma unroll
  for (int j = 0; j < 4; ++j) {
    int n = nBlk + waveN * 64 + j * 16 + lrow;
    float bv = bias ? bias[n] : 0.f;
#pragma unroll
    for (int i = 0; i < 2; ++i) {
#pragma unroll
      for (int r = 0; r < 8; ++r) {
        int m = mBlk + waveM * 32 + i * 16 + kh * 8 + r;
        float v = acc[i][j][r] + bv;
        if (act) v = v / (1.f + __expf(-v));   // SiLU
        size_t idx = (size_t)m * (size_t)N + (size_t)n;
        if (resid) v += resid[idx];
        if (outF) outF[idx] = v;
        if (outH) outH[idx] = (bf16_t)v;
      }
    }
  }
}

// ---------------------------------------------------------------------------
// Host orchestration
// ---------------------------------------------------------------------------
extern "C" void kernel_launch(void* const* d_in, const int* in_sizes, int n_in,
                              void* d_out, int out_size, void* d_ws, size_t ws_size,
                              hipStream_t stream) {
  const int*   tokens   = (const int*)d_in[0];
  const float* embed    = (const float*)d_in[1];
  const float* ln1_w    = (const float*)d_in[2];
  const float* ln1_b    = (const float*)d_in[3];
  const float* fc_w     = (const float*)d_in[4];
  const float* fc_b     = (const float*)d_in[5];
  const float* ph_init  = (const float*)d_in[6];
  const float* ph_angle = (const float*)d_in[7];
  const float* lnmln    = (const float*)d_in[8];
  const float* lastc    = (const float*)d_in[9];
  const float* ln2_w    = (const float*)d_in[10];
  const float* ln2_b    = (const float*)d_in[11];
  const float* f1_w     = (const float*)d_in[12];
  const float* f1_b     = (const float*)d_in[13];
  const float* f2_w     = (const float*)d_in[14];
  const float* f2_b     = (const float*)d_in[15];
  const float* lnf_w    = (const float*)d_in[16];
  const float* lnf_b    = (const float*)d_in[17];
  const float* out_w    = (const float*)d_in[18];
  const float* out_b    = (const float*)d_in[19];
  float* logits = (float*)d_out;

  char* ws = (char*)d_ws;
  size_t off = 0;
  auto carve = [&](size_t bytes) -> void* {
    void* p = ws + off;
    off += (bytes + 255) & ~(size_t)255;
    return p;
  };

  float*  x     = (float*)carve((size_t)MTOK * DIMN * 4);
  float*  xn    = (float*)carve((size_t)MTOK * DIMN * 4);
  float*  gate  = (float*)carve((size_t)MTOK * DIMN * 4);
  bf16_t* xnh   = (bf16_t*)carve((size_t)MTOK * DIMN * 2);
  bf16_t* hh    = (bf16_t*)carve((size_t)MTOK * FFN * 2);
  float2* cs    = (float2*)carve((size_t)BBN * DIMN * NCH * 8);
  bf16_t* fcw_h = (bf16_t*)carve((size_t)DIMN * DIMN * 2);
  bf16_t* f1w_h = (bf16_t*)carve((size_t)FFN * DIMN * 2);
  bf16_t* f2w_h = (bf16_t*)carve((size_t)DIMN * FFN * 2);
  bf16_t* outw_h= (bf16_t*)carve((size_t)VOCABN * DIMN * 2);

  // Embedding + max-norm clip
  k_embed<<<MTOK, 256, 0, stream>>>(tokens, embed, x);

  const int sconvBlocks = (BBN * NCH * DIMN + 255) / 256;  // 128

  for (int l = 0; l < DEPTHN; ++l) {
    const float* lfc  = fc_w + (size_t)l * DIMN * DIMN;
    const float* lf1  = f1_w + (size_t)l * FFN * DIMN;
    const float* lf2  = f2_w + (size_t)l * DIMN * FFN;

    // LN1 -> xn (fp32 for sconv) + xnh (bf16 for GEMM)
    k_layernorm<<<MTOK, 256, 0, stream>>>(x, ln1_w + l * DIMN, ln1_b + l * DIMN, xn, xnh);

    // gate = silu(xn @ fc^T + fc_b)
    k_cvt_bf16<<<2048, 256, 0, stream>>>(lfc, fcw_h, (size_t)DIMN * DIMN);
    k_gemm_bf16<<<dim3(MTOK / 128, DIMN / 128), 256, 0, stream>>>(
        xnh, fcw_h, fc_b + l * DIMN, nullptr, gate, nullptr, DIMN, DIMN, 1);

    // sconv (chunked complex scan), fused with gate multiply + residual into x
    k_sconv_scan<<<sconvBlocks, 256, 0, stream>>>(xn, ph_angle + l * DIMN, lnmln + l * DIMN, cs);
    k_sconv_combine<<<(BBN * DIMN + 255) / 256, 256, 0, stream>>>(
        ph_angle + l * DIMN, lnmln + l * DIMN, cs);
    k_sconv_out<<<sconvBlocks, 256, 0, stream>>>(
        xn, gate, x, ph_angle + l * DIMN, lnmln + l * DIMN,
        ph_init + (size_t)l * DIMN * 2, lastc + (size_t)l * DIMN * 2, cs);

    // LN2
    k_layernorm<<<MTOK, 256, 0, stream>>>(x, ln2_w + l * DIMN, ln2_b + l * DIMN, xn, xnh);

    // h = silu(xn @ f1^T + f1_b)  (bf16 out, feeds next GEMM)
    k_cvt_bf16<<<2048, 256, 0, stream>>>(lf1, f1w_h, (size_t)FFN * DIMN);
    k_gemm_bf16<<<dim3(MTOK / 128, FFN / 128), 256, 0, stream>>>(
        xnh, f1w_h, f1_b + l * FFN, nullptr, nullptr, hh, FFN, DIMN, 1);

    // x = h @ f2^T + f2_b + x   (residual fused)
    k_cvt_bf16<<<2048, 256, 0, stream>>>(lf2, f2w_h, (size_t)DIMN * FFN);
    k_gemm_bf16<<<dim3(MTOK / 128, DIMN / 128), 256, 0, stream>>>(
        hh, f2w_h, f2_b + l * DIMN, x, x, nullptr, DIMN, FFN, 0);
  }

  // Final LN + vocab projection
  k_layernorm<<<MTOK, 256, 0, stream>>>(x, lnf_w, lnf_b, xn, xnh);
  k_cvt_bf16<<<4096, 256, 0, stream>>>(out_w, outw_h, (size_t)VOCABN * DIMN);
  k_gemm_bf16<<<dim3(MTOK / 128, VOCABN / 128), 256, 0, stream>>>(
      xnh, outw_h, out_b, nullptr, logits, nullptr, VOCABN, DIMN, 0);
}
